// MatrixFactorization_37228776521760
// MI455X (gfx1250) — matrix operand order repaired and checked by reading, not compiled, not run on hardware
//
#include <hip/hip_runtime.h>

typedef __attribute__((ext_vector_type(16))) _Float16 v16h;
typedef __attribute__((ext_vector_type(8)))  _Float16 v8h;
typedef __attribute__((ext_vector_type(8)))  float    v8f;

#define MF_BATCH 16384
#define MF_EMB   64
#define MF_HID   128
#define MF_ROWS  64      // batch rows per block
#define MF_THR   128     // 4 waves of 32
#define MF_LDA   136     // padded LDS row stride in halves (128 + 8) -> conflict-free

union FragU { v16h v; v8h h[2]; };

__global__ __launch_bounds__(MF_THR)
void mf_fused_kernel(const int*   __restrict__ users,
                     const int*   __restrict__ items,
                     const float* __restrict__ user_emb,
                     const float* __restrict__ item_emb,
                     const float* __restrict__ user_bias,
                     const float* __restrict__ item_bias,
                     const float* __restrict__ global_bias,
                     const float* __restrict__ w1,
                     const float* __restrict__ b1,
                     const float* __restrict__ w2,
                     const float* __restrict__ b2,
                     float*       __restrict__ out)
{
    __shared__ __align__(16) _Float16 sA[MF_ROWS * MF_LDA];  // mlp_in tile (f16)
    __shared__ __align__(16) _Float16 sW[MF_HID * MF_LDA];   // w1 (f16), row n = hidden unit
    __shared__ float sBase[MF_ROWS];                         // dot + ub + ib + gb + b2

    const int tid  = threadIdx.x;
    const int row0 = blockIdx.x * MF_ROWS;

    // ---- Phase 1a: stage w1 fp32 -> f16 into LDS (coalesced float4 reads) ----
    #pragma unroll
    for (int itr = 0; itr < 32; ++itr) {
        int e = (itr * MF_THR + tid) * 4;          // element index in [0, 16384)
        float4 wv = *(const float4*)(w1 + e);
        int n = e >> 7;                            // hidden-unit row
        int k = e & 127;                           // input column
        _Float16* dst = &sW[n * MF_LDA + k];
        dst[0] = (_Float16)wv.x; dst[1] = (_Float16)wv.y;
        dst[2] = (_Float16)wv.z; dst[3] = (_Float16)wv.w;
    }

    // ---- Phase 1b: gather embeddings, fp32 dot, stage mlp_in as f16 ----
    {
        int r = tid >> 1;                          // local row 0..63
        int h = tid & 1;                           // half of the 64-dim embedding
        int u = users[row0 + r];
        int v = items[row0 + r];
        const float* up = user_emb + (size_t)u * MF_EMB + h * 32;
        const float* vp = item_emb + (size_t)v * MF_EMB + h * 32;
        _Float16* au = &sA[r * MF_LDA + h * 32];
        _Float16* av = &sA[r * MF_LDA + MF_EMB + h * 32];
        float dotp = 0.f;
        #pragma unroll
        for (int i = 0; i < 8; ++i) {
            float4 uu = *(const float4*)(up + i * 4);
            float4 vv = *(const float4*)(vp + i * 4);
            dotp += uu.x * vv.x + uu.y * vv.y + uu.z * vv.z + uu.w * vv.w;
            au[i*4+0] = (_Float16)uu.x; au[i*4+1] = (_Float16)uu.y;
            au[i*4+2] = (_Float16)uu.z; au[i*4+3] = (_Float16)uu.w;
            av[i*4+0] = (_Float16)vv.x; av[i*4+1] = (_Float16)vv.y;
            av[i*4+2] = (_Float16)vv.z; av[i*4+3] = (_Float16)vv.w;
        }
        dotp += __shfl_xor(dotp, 1, 32);           // pair reduce (same wave)
        if (h == 0) {
            sBase[r] = dotp + user_bias[u] + item_bias[v] + global_bias[0] + b2[0];
        }
    }
    __syncthreads();

    // ---- Phase 2: [16 x 128] @ [128 x 128] via v_wmma_f32_16x16x32_f16 ----
    const int lane = tid & 31;
    const int wave = tid >> 5;        // 0..3 -> M-tile of 16 rows
    const int col  = lane & 15;       // N / M position within tile
    const int hs   = lane >> 4;       // half-wave select

    // A fragments for all 4 K-steps (ISA 16-bit A layout: two 8-half chunks/lane)
    v16h afrag[4];
    {
        const _Float16* arow = &sA[(wave * 16 + col) * MF_LDA];
        #pragma unroll
        for (int kk = 0; kk < 4; ++kk) {
            FragU t;
            t.h[0] = *(const v8h*)(arow + kk * 32 + hs * 8);
            t.h[1] = *(const v8h*)(arow + kk * 32 + 16 + hs * 8);
            afrag[kk] = t.v;
        }
    }

    // Accumulators for 8 N-tiles, seeded with b1[N] (bias applies to every row)
    v8f acc[8];
    #pragma unroll
    for (int nt = 0; nt < 8; ++nt) {
        float bias = b1[nt * 16 + col];
        #pragma unroll
        for (int i = 0; i < 8; ++i) acc[nt][i] = bias;
    }

    #pragma unroll
    for (int kk = 0; kk < 4; ++kk) {
        #pragma unroll
        for (int nt = 0; nt < 8; ++nt) {
            // B fragment: lane holds column n = nt*16+col of w1^T over its K range,
            // i.e. two 8-half chunks of w1 row n (k = 8*hs.., 16+8*hs..), matching A.
            const _Float16* wrow = &sW[(nt * 16 + col) * MF_LDA + kk * 32 + hs * 8];
            FragU b;
            b.h[0] = *(const v8h*)(wrow);
            b.h[1] = *(const v8h*)(wrow + 16);
            acc[nt] = __builtin_amdgcn_wmma_f32_16x16x32_f16(
                false, afrag[kk], false, b.v, (short)0, acc[nt], false, false);
        }
    }

    // ---- Epilogue: ReLU, h @ w2 (N=1) as lane reduction, add base, store ----
    float p[8];
    #pragma unroll
    for (int r = 0; r < 8; ++r) p[r] = 0.f;

    #pragma unroll
    for (int nt = 0; nt < 8; ++nt) {
        float w2v = w2[nt * 16 + col];
        #pragma unroll
        for (int r = 0; r < 8; ++r) {
            float hv = acc[nt][r];
            hv = hv > 0.f ? hv : 0.f;              // ReLU
            p[r] += hv * w2v;
        }
    }
    // reduce across the 16 lanes of each half-wave (masks < 16 stay in-half)
    #pragma unroll
    for (int m = 1; m < 16; m <<= 1) {
        #pragma unroll
        for (int r = 0; r < 8; ++r) p[r] += __shfl_xor(p[r], m, 32);
    }

    if (col == 0) {
        #pragma unroll
        for (int r = 0; r < 8; ++r) {
            int rl = wave * 16 + hs * 8 + r;       // C layout: VGPR r -> row r + 8*hs
            out[row0 + rl] = sBase[rl] + p[r];
        }
    }
}

extern "C" void kernel_launch(void* const* d_in, const int* in_sizes, int n_in,
                              void* d_out, int out_size, void* d_ws, size_t ws_size,
                              hipStream_t stream) {
    (void)in_sizes; (void)n_in; (void)d_ws; (void)ws_size; (void)out_size;
    const int*   users       = (const int*)  d_in[0];
    const int*   items       = (const int*)  d_in[1];
    const float* user_emb    = (const float*)d_in[2];
    const float* item_emb    = (const float*)d_in[3];
    const float* user_bias   = (const float*)d_in[4];
    const float* item_bias   = (const float*)d_in[5];
    const float* global_bias = (const float*)d_in[6];
    const float* w1          = (const float*)d_in[7];
    const float* b1          = (const float*)d_in[8];
    const float* w2          = (const float*)d_in[9];
    const float* b2          = (const float*)d_in[10];
    float*       out         = (float*)d_out;

    dim3 grid(MF_BATCH / MF_ROWS);   // 256 blocks
    dim3 block(MF_THR);              // 128 threads = 4 waves (wave32)
    mf_fused_kernel<<<grid, block, 0, stream>>>(
        users, items, user_emb, item_emb, user_bias, item_bias,
        global_bias, w1, b1, w2, b2, out);
}
